// SelfAttention_31920196944294
// MI455X (gfx1250) — compile-verified, matrix operand order checked by reading, not run on hardware
//
#include <hip/hip_runtime.h>
#include <hip/hip_bf16.h>
#include <cstdint>

// Self-attention forward for MI455X (gfx1250, wave32, WMMA + async TDM).
// fp32 inputs are down-converted to bf16; all GEMM work runs on
// v_wmma_f32_16x16x32_bf16 with fp32 accumulation. Attention is flash-style
// with online softmax; K tiles are double-buffered in LDS via
// TENSOR_LOAD_TO_LDS with a pipelined s_wait_tensorcnt(1).

typedef __attribute__((ext_vector_type(16))) __bf16   bf16x16;
typedef __attribute__((ext_vector_type(4)))  __bf16   bf16x4;
typedef __attribute__((ext_vector_type(8)))  float    f32x8;
typedef __attribute__((ext_vector_type(4)))  unsigned u32x4;
typedef __attribute__((ext_vector_type(8)))  int      i32x8;
typedef __attribute__((ext_vector_type(4)))  int      i32x4;

#define DIM       1024
#define THREE_DIM 3072
#define NHEADS    16
#define HDIM      64
#define SEQ       2048
#define BATCH     4
#define ROWS      (BATCH * SEQ)          // 8192
#define ATTN_SCALE 0.125f                // 64^-0.5

#if defined(__HIP_DEVICE_COMPILE__) && \
    __has_builtin(__builtin_amdgcn_tensor_load_to_lds) && \
    __has_builtin(__builtin_amdgcn_s_wait_tensorcnt)
#define USE_TDM 1
#else
#define USE_TDM 0
#endif

// ---------------------------------------------------------------------------
// WMMA fragment loader (CDNA5 ISA 7.12.2, wave32).
// A fragment 16(M)x32(K), 16-bit: lane = m + 16*h
//   elems 0..7  -> K = koff + h*8 + i        (contiguous 16B run)
//   elems 8..15 -> K = koff + 16 + h*8 + i-8 (contiguous 16B run)
// A B-fragment of a matrix stored transposed ([N,K] row-major) has the same
// addressing with lane = n + 16*h, so this one loader serves both operands.
// ---------------------------------------------------------------------------
__device__ __forceinline__ bf16x16 load_frag_a(const __bf16* p, int lda,
                                               int koff, int lane) {
  const int m = lane & 15;
  const int h = lane >> 4;
  const __bf16* row = p + (long)m * lda;
  bf16x16 r;
#pragma unroll
  for (int i = 0; i < 8; ++i) r[i] = row[koff + h * 8 + i];
#pragma unroll
  for (int i = 0; i < 8; ++i) r[8 + i] = row[koff + 16 + h * 8 + i];
  return r;
}

__device__ __forceinline__ f32x8 wmma_bf16(bf16x16 a, bf16x16 b, f32x8 c) {
  return __builtin_amdgcn_wmma_f32_16x16x32_bf16(
      false, a, false, b, (short)0, c, false, false);
}

// ---------------------------------------------------------------------------
// TDM: 2D tile load Global -> LDS (ISA ch.8 D# descriptor, §8.3/8.4)
// ---------------------------------------------------------------------------
#if USE_TDM
__device__ __forceinline__ void tdm_load_tile_2d(
    const __bf16* gsrc, unsigned lds_byte_addr,
    int tile_rows, int tile_cols, int row_stride_elems, int tensor_rows) {
  const unsigned long long ga = (unsigned long long)(uintptr_t)gsrc;
  u32x4 g0;
  g0[0] = 1u;                                        // count=1, user desc
  g0[1] = lds_byte_addr;                             // lds_addr [63:32]
  g0[2] = (unsigned)(ga & 0xffffffffu);              // global_addr low
  g0[3] = (unsigned)((ga >> 32) & 0x01ffffffu)       // global_addr high
        | (2u << 30);                                // type=2 (image)
  i32x8 g1;
  g1[0] = 1 << 16;                                   // data_size=1 (2 bytes)
  g1[1] = (tile_cols & 0xffff) << 16;                // tensor_dim0 lo16
  g1[2] = ((tile_cols >> 16) & 0xffff)               // tensor_dim0 hi16
        | ((tensor_rows & 0xffff) << 16);            // tensor_dim1 lo16
  g1[3] = ((tensor_rows >> 16) & 0xffff)             // tensor_dim1 hi16
        | ((tile_cols & 0xffff) << 16);              // tile_dim0
  g1[4] = tile_rows & 0xffff;                        // tile_dim1 (tile_dim2=0)
  g1[5] = row_stride_elems;                          // tensor_dim0_stride lo32
  g1[6] = 0;                                         // stride hi / dim1_stride
  g1[7] = 0;
  const i32x4 gz = {0, 0, 0, 0};
#if __clang_major__ >= 23
  const i32x8 gz8 = {0, 0, 0, 0, 0, 0, 0, 0};
  __builtin_amdgcn_tensor_load_to_lds(g0, g1, gz, gz, gz8, 0);
#else
  __builtin_amdgcn_tensor_load_to_lds(g0, g1, gz, gz, 0);
#endif
}
#endif

// ---------------------------------------------------------------------------
// fp32 -> bf16 conversion, 4 elements/thread (pure HBM streaming)
// ---------------------------------------------------------------------------
__global__ void cvt_f32_to_bf16_v4(const float4* __restrict__ in,
                                   bf16x4* __restrict__ out, int n4) {
  int i = blockIdx.x * blockDim.x + threadIdx.x;
  const int stride = gridDim.x * blockDim.x;
  for (; i < n4; i += stride) {
    const float4 f = in[i];
    bf16x4 o;
    o[0] = (__bf16)f.x; o[1] = (__bf16)f.y;
    o[2] = (__bf16)f.z; o[3] = (__bf16)f.w;
    out[i] = o;
  }
}

// fp32 [K,N] -> bf16 [N,K]; blockIdx.x = source row k (no div/mod).
__global__ void cvt_transpose_bf16(const float* __restrict__ in,
                                   __bf16* __restrict__ out, int K, int N) {
  const int k = blockIdx.x;
  const float* src = in + (long)k * N;
  for (int n = blockIdx.y * blockDim.x + threadIdx.x; n < N;
       n += gridDim.y * blockDim.x)
    out[(long)n * K + k] = (__bf16)src[n];
}

// ---------------------------------------------------------------------------
// bf16 GEMM: C[M,N] = A[M,K] @ B[K,N] + bias, with B supplied transposed
// (BT = [N,K] row-major). One 32x64 macro-tile per wave: 2 A frags x 4 B
// frags = 8 WMMA per 32-deep K step; every fragment load is 2x b128.
// ---------------------------------------------------------------------------
template <bool BF16_OUT>
__global__ __launch_bounds__(128) void gemm_bf16_wmma(
    const __bf16* __restrict__ A, const __bf16* __restrict__ BT,
    const float* __restrict__ bias, void* __restrict__ outp,
    int M, int N, int K) {
  const int lane = threadIdx.x & 31;
  const int wid  = blockIdx.x * (blockDim.x >> 5) + (threadIdx.x >> 5);
  const int tiles_n = N >> 6;                  // 64-col tiles
  const int tm = wid / tiles_n;                // 32-row tile index
  const int tn = wid % tiles_n;
  if (tm * 32 >= M) return;

  const __bf16* a0p = A + (long)tm * 32 * K;
  const __bf16* a1p = a0p + (long)16 * K;
  const __bf16* btp = BT + (long)tn * 64 * K;

  f32x8 acc[2][4] = {};
  for (int k = 0; k < K; k += 32) {
    __builtin_prefetch(a0p + ((lane & 15) * (long)K) + k + 128, 0, 0);
    __builtin_prefetch(btp + ((lane & 15) * (long)K) + k + 128, 0, 0);
    const bf16x16 a0 = load_frag_a(a0p, K, k, lane);
    const bf16x16 a1 = load_frag_a(a1p, K, k, lane);
#pragma unroll
    for (int c = 0; c < 4; ++c) {
      const bf16x16 b = load_frag_a(btp + (long)c * 16 * K, K, k, lane);
      acc[0][c] = wmma_bf16(a0, b, acc[0][c]);
      acc[1][c] = wmma_bf16(a1, b, acc[1][c]);
    }
  }

  const int half = lane >> 4;
  const int nloc = lane & 15;
#pragma unroll
  for (int rt = 0; rt < 2; ++rt) {
#pragma unroll
    for (int c = 0; c < 4; ++c) {
      const int col = tn * 64 + c * 16 + nloc;
      const float bv = bias[col];
#pragma unroll
      for (int v = 0; v < 8; ++v) {
        const int row = tm * 32 + rt * 16 + v + 8 * half;  // D layout
        const float val = acc[rt][c][v] + bv;
        if (BF16_OUT) ((__bf16*)outp)[(long)row * N + col] = (__bf16)val;
        else          ((float*)outp)[(long)row * N + col]  = val;
      }
    }
  }
}

// ---------------------------------------------------------------------------
// Flash attention: one workgroup (4 waves) per (b, h, 64 query rows).
// qkv layout [B, N, 3072]: q at col h*64, k at 1024+h*64, v at 2048+h*64.
// K tiles double-buffered in LDS via async TDM (wave 0 issues, TENSORcnt=1
// stays in flight); V tile stored transposed in LDS so PV B-fragments are
// contiguous ds_load_b128s.
// ---------------------------------------------------------------------------
__global__ __launch_bounds__(128) void attn_fwd_wmma(
    const __bf16* __restrict__ qkv, __bf16* __restrict__ attn_out) {
  const int blk = blockIdx.x;          // B * H * (N/64) = 2048
  const int qchunk = blk & 31;
  const int h = (blk >> 5) & 15;
  const int b = blk >> 9;
  const int w    = threadIdx.x >> 5;
  const int lane = threadIdx.x & 31;
  const int q0   = qchunk * 64 + w * 16;

  __shared__ __align__(16) __bf16 sK[2][32 * HDIM];  // double-buffered K tile
  __shared__ __align__(16) __bf16 sVT[HDIM * 32];    // V transposed: [d][key]
  __shared__ __align__(16) __bf16 sP[4][16 * 32];    // per-wave P relayout

  const __bf16* qptr = qkv + ((long)b * SEQ) * THREE_DIM + h * HDIM;
  const __bf16* kptr = qptr + DIM;
  const __bf16* vptr = qptr + 2 * DIM;

  const __bf16* qrow = qptr + (long)q0 * THREE_DIM;
  const bf16x16 qa0 = load_frag_a(qrow, THREE_DIM, 0,  lane);
  const bf16x16 qa1 = load_frag_a(qrow, THREE_DIM, 32, lane);

  float m_run[8], l_run[8];
#pragma unroll
  for (int v = 0; v < 8; ++v) { m_run[v] = -1e30f; l_run[v] = 0.0f; }
  f32x8 o0 = {}, o1 = {}, o2 = {}, o3 = {};

  const int t  = threadIdx.x;
  const int lr = t >> 3;               // 0..15: key row within half-tile
  const int lp = t & 7;                // 16B chunk within 128B row

#if USE_TDM
  if (w == 0)                          // prologue: start tile 0
    tdm_load_tile_2d(kptr, (unsigned)(uintptr_t)(void*)&sK[0][0],
                     32, HDIM, THREE_DIM, SEQ);
#endif

  for (int kb = 0; kb < SEQ; kb += 32) {
    const int buf = (kb >> 5) & 1;
    const __bf16* sKc = &sK[buf][0];

    // ---- K tile -> LDS (pipelined TDM: keep next tile in flight) ----
#if USE_TDM
    if (w == 0) {
      if (kb + 32 < SEQ) {
        tdm_load_tile_2d(kptr + (long)(kb + 32) * THREE_DIM,
                         (unsigned)(uintptr_t)(void*)&sK[buf ^ 1][0],
                         32, HDIM, THREE_DIM, SEQ);
        __builtin_amdgcn_s_wait_tensorcnt(1);   // current tile complete
      } else {
        __builtin_amdgcn_s_wait_tensorcnt(0);
      }
    }
#else
    {
      const uint4* ks0 = (const uint4*)(kptr + (long)(kb + lr) * THREE_DIM) + lp;
      const uint4* ks1 = (const uint4*)(kptr + (long)(kb + 16 + lr) * THREE_DIM) + lp;
      ((uint4*)sKc)[t]       = *ks0;
      ((uint4*)sKc)[t + 128] = *ks1;
    }
#endif
    // ---- V tile -> LDS, transposed ([d][key]) ----
    {
      const uint4 vv0 = *((const uint4*)(vptr + (long)(kb + lr) * THREE_DIM) + lp);
      const uint4 vv1 = *((const uint4*)(vptr + (long)(kb + 16 + lr) * THREE_DIM) + lp);
      const __bf16* e0 = (const __bf16*)&vv0;
      const __bf16* e1 = (const __bf16*)&vv1;
#pragma unroll
      for (int i = 0; i < 8; ++i) {
        sVT[(lp * 8 + i) * 32 + lr]      = e0[i];
        sVT[(lp * 8 + i) * 32 + lr + 16] = e1[i];
      }
    }
    __syncthreads();

    // ---- S = Q @ K^T (two 16x16 score tiles over 32 keys) ----
    const bf16x16 kf00 = load_frag_a(sKc,             HDIM, 0,  lane);
    const bf16x16 kf01 = load_frag_a(sKc,             HDIM, 32, lane);
    const bf16x16 kf10 = load_frag_a(sKc + 16 * HDIM, HDIM, 0,  lane);
    const bf16x16 kf11 = load_frag_a(sKc + 16 * HDIM, HDIM, 32, lane);
    f32x8 sa = {}, sb = {};
    sa = wmma_bf16(qa0, kf00, sa);
    sa = wmma_bf16(qa1, kf01, sa);
    sb = wmma_bf16(qa0, kf10, sb);
    sb = wmma_bf16(qa1, kf11, sb);

    // ---- online softmax (rows live in 16-lane halves) ----
#pragma unroll
    for (int v = 0; v < 8; ++v) {
      const float sav = sa[v] * ATTN_SCALE;
      const float sbv = sb[v] * ATTN_SCALE;
      float mx = fmaxf(sav, sbv);
      mx = fmaxf(mx, __shfl_xor(mx, 1, 32));
      mx = fmaxf(mx, __shfl_xor(mx, 2, 32));
      mx = fmaxf(mx, __shfl_xor(mx, 4, 32));
      mx = fmaxf(mx, __shfl_xor(mx, 8, 32));
      const float mn = fmaxf(m_run[v], mx);
      const float alpha = __expf(m_run[v] - mn);
      const float ea = __expf(sav - mn);
      const float eb = __expf(sbv - mn);
      float rs = ea + eb;
      rs += __shfl_xor(rs, 1, 32);
      rs += __shfl_xor(rs, 2, 32);
      rs += __shfl_xor(rs, 4, 32);
      rs += __shfl_xor(rs, 8, 32);
      l_run[v] = l_run[v] * alpha + rs;
      m_run[v] = mn;
      o0[v] *= alpha; o1[v] *= alpha; o2[v] *= alpha; o3[v] *= alpha;
      const int rr = v + 8 * (lane >> 4);
      sP[w][rr * 32 + (lane & 15)]      = (__bf16)ea;
      sP[w][rr * 32 + 16 + (lane & 15)] = (__bf16)eb;
    }
    __syncthreads();

    // ---- O += P @ V; V B-frags come from transposed LDS (contiguous) ----
    const bf16x16 pf  = load_frag_a(sP[w], 32, 0, lane);
    const bf16x16 vf0 = load_frag_a(sVT + 0 * 16 * 32, 32, 0, lane);
    const bf16x16 vf1 = load_frag_a(sVT + 1 * 16 * 32, 32, 0, lane);
    const bf16x16 vf2 = load_frag_a(sVT + 2 * 16 * 32, 32, 0, lane);
    const bf16x16 vf3 = load_frag_a(sVT + 3 * 16 * 32, 32, 0, lane);
    o0 = wmma_bf16(pf, vf0, o0);
    o1 = wmma_bf16(pf, vf1, o1);
    o2 = wmma_bf16(pf, vf2, o2);
    o3 = wmma_bf16(pf, vf3, o3);
    __syncthreads();
  }

  // ---- normalize and write [B, N, 1024] bf16 ----
#pragma unroll
  for (int v = 0; v < 8; ++v) {
    const float inv = 1.0f / l_run[v];
    const int n = q0 + v + 8 * (lane >> 4);
    const long base = ((long)b * SEQ + n) * DIM + h * HDIM;
    attn_out[base + 0  + (lane & 15)] = (__bf16)(o0[v] * inv);
    attn_out[base + 16 + (lane & 15)] = (__bf16)(o1[v] * inv);
    attn_out[base + 32 + (lane & 15)] = (__bf16)(o2[v] * inv);
    attn_out[base + 48 + (lane & 15)] = (__bf16)(o3[v] * inv);
  }
}

// ---------------------------------------------------------------------------
// Launch
// ---------------------------------------------------------------------------
extern "C" void kernel_launch(void* const* d_in, const int* in_sizes, int n_in,
                              void* d_out, int out_size, void* d_ws, size_t ws_size,
                              hipStream_t stream) {
  const float* x      = (const float*)d_in[0];   // [8192, 1024]
  const float* W_qkv  = (const float*)d_in[1];   // [1024, 3072]
  const float* b_qkv  = (const float*)d_in[2];   // [3072]
  const float* W_proj = (const float*)d_in[3];   // [1024, 1024]
  const float* b_proj = (const float*)d_in[4];   // [1024]
  float* out = (float*)d_out;                    // [8192, 1024]

  char* ws = (char*)d_ws;
  __bf16* xb     = (__bf16*)(ws);                                  // 16 MB
  __bf16* wqkvT  = (__bf16*)(ws + (size_t)16 * 1024 * 1024);       // 6 MB  [3072,1024]
  __bf16* qkvb   = (__bf16*)(ws + (size_t)23 * 1024 * 1024);       // 48 MB [B,N,3072]
  __bf16* attnb  = (__bf16*)(ws + (size_t)72 * 1024 * 1024);       // 16 MB [B,N,1024]
  __bf16* wprjT  = (__bf16*)(ws + (size_t)89 * 1024 * 1024);       // 2 MB  [1024,1024]

  // 1) conversions (weights transposed for contiguous B-fragment loads)
  cvt_f32_to_bf16_v4<<<2048, 256, 0, stream>>>(
      (const float4*)x, (bf16x4*)xb, ROWS * DIM / 4);
  cvt_transpose_bf16<<<dim3(DIM, 12), 256, 0, stream>>>(W_qkv, wqkvT, DIM, THREE_DIM);
  cvt_transpose_bf16<<<dim3(DIM, 4), 256, 0, stream>>>(W_proj, wprjT, DIM, DIM);

  // 2) QKV GEMM: [8192,1024] @ [1024,3072] + b_qkv -> bf16 [8192,3072]
  {
    const int waves = (ROWS / 32) * (THREE_DIM / 64);   // 256*48 = 12288
    gemm_bf16_wmma<true><<<waves / 4, 128, 0, stream>>>(
        xb, wqkvT, b_qkv, qkvb, ROWS, THREE_DIM, DIM);
  }

  // 3) Flash attention per (b, h, 64-row query block)
  attn_fwd_wmma<<<BATCH * NHEADS * (SEQ / 64), 128, 0, stream>>>(qkvb, attnb);

  // 4) Projection GEMM: [8192,1024] @ [1024,1024] + b_proj -> fp32 out
  {
    const int waves = (ROWS / 32) * (DIM / 64);         // 256*16 = 4096
    gemm_bf16_wmma<false><<<waves / 4, 128, 0, stream>>>(
        attnb, wprjT, b_proj, out, ROWS, DIM, DIM);
  }
}
